// SFB2_52982716563839
// MI455X (gfx1250) — matrix operand order repaired and checked by reading, not compiled, not run on hardware
//
#include <hip/hip_runtime.h>
#include <hip/hip_bf16.h>

typedef __attribute__((ext_vector_type(16))) _Float16 v16h;
typedef __attribute__((ext_vector_type(8)))  float    v8f;

#define HW   65536   // 256*256
#define ND   4096    // 64*64 downsampled pixels per batch
#define NB   4
#define NC   64
#define THITA 1e-4f

static __device__ __forceinline__ v8f v8f_zero() {
  v8f z;
#pragma unroll
  for (int e = 0; e < 8; ++e) z[e] = 0.f;
  return z;
}

// WMMA accumulator = A(16x32 f16) x B(32x16 f16) + C(16x16 f32)
static __device__ __forceinline__ v8f wmma(v16h a, v16h b, v8f c) {
  return __builtin_amdgcn_wmma_f32_16x16x32_f16(false, a, false, b, (short)0, c,
                                                false, false);
}

// ---- A-matrix (16x32, f16) fragment swizzle -------------------------------
// lane L: M = L&15, hi = L>>4 ; halfword h: g=h>>3, j=h&7, K = g*16 + hi*8 + j
// Weights [O x K] stored in LDS so that the fragment for (chunk, mtile) is 16
// contiguous halfwords per lane (two ds_load_b128 per fragment).
static __device__ __forceinline__ int wswz(int o, int i) {
  int m = o >> 4, M = o & 15;
  int chunk = i >> 5, kl = i & 31;
  int g = kl >> 4, rem = kl & 15, hi = rem >> 3, j = kl & 7;
  int lane = M + (hi << 4);
  int h = (g << 3) + j;
  return ((chunk * 4 + m) * 32 + lane) * 16 + h;   // 4 M-tiles (64 out ch)
}

// Build B fragment (32x16) from f32 global rows: elem(k,n) = src[k*stride+px]
// lane L: N = L&15 (caller folds N into px), halfword h: K = (h>>3)*16 + (L>>4)*8 + (h&7).
static __device__ __forceinline__ v16h bfrag_f32(const float* __restrict__ src,
                                                 size_t stride, size_t px,
                                                 int hi) {
  v16h b;
#pragma unroll
  for (int h = 0; h < 16; ++h)
    b[h] = (_Float16)src[(size_t)((h >> 3) * 16 + hi * 8 + (h & 7)) * stride + px];
  return b;
}
static __device__ __forceinline__ v16h bfrag_f16(const _Float16* __restrict__ src,
                                                 size_t stride, size_t px,
                                                 int hi) {
  v16h b;
#pragma unroll
  for (int h = 0; h < 16; ++h)
    b[h] = src[(size_t)((h >> 3) * 16 + hi * 8 + (h & 7)) * stride + px];
  return b;
}
// B fragment from per-wave LDS buffer laid out as [k][n] f16 (stride 16).
static __device__ __forceinline__ v16h bfrag_lds(const _Float16* buf, int kbase,
                                                 int n, int hi) {
  v16h b;
#pragma unroll
  for (int h = 0; h < 16; ++h) b[h] = buf[(kbase + (h >> 3) * 16 + hi * 8 + (h & 7)) * 16 + n];
  return b;
}

// ===========================================================================
// Kernel A: x = ReLU(Wx0[64x192] * [x1;x2] + bx0)  full resolution -> d_out
// One wave = 32 pixels x 64 out channels (two N-groups share each A frag).
// K = 192 (6 chunks of 32).
// ===========================================================================
__global__ void __launch_bounds__(128)
k_convx(const float* __restrict__ x1, const float* __restrict__ x2,
        const float* __restrict__ w, const float* __restrict__ bias,
        float* __restrict__ out) {
  __shared__ __align__(32) _Float16 wls[6 * 4 * 32 * 16];  // 12288 f16
  __shared__ float bls[64];
  const int tid = threadIdx.x;
  for (int t = tid; t < 64 * 192; t += 128)
    wls[wswz(t / 192, t % 192)] = (_Float16)w[t];
  if (tid < 64) bls[tid] = bias[tid];
  __syncthreads();

  const int lane = tid & 31, wv = tid >> 5;
  const int n = lane & 15, hi = lane >> 4;
  const int b = blockIdx.y;
  const int p0 = blockIdx.x * 128 + wv * 32;
  const size_t px0 = (size_t)p0 + n;
  const size_t px1 = (size_t)p0 + 16 + n;

  v8f acc0[4] = {v8f_zero(), v8f_zero(), v8f_zero(), v8f_zero()};
  v8f acc1[4] = {v8f_zero(), v8f_zero(), v8f_zero(), v8f_zero()};
#pragma unroll
  for (int chunk = 0; chunk < 6; ++chunk) {
    const float* src = (chunk < 4)
        ? (x1 + (size_t)b * 128 * HW + (size_t)chunk * 32 * HW)
        : (x2 + (size_t)b * 64 * HW + (size_t)(chunk - 4) * 32 * HW);
    if (chunk < 5) {  // warm L2 for the next K-chunk (global_prefetch_b8)
      const float* nsrc = (chunk + 1 < 4)
          ? (x1 + (size_t)b * 128 * HW + (size_t)(chunk + 1) * 32 * HW)
          : (x2 + (size_t)b * 64 * HW + (size_t)(chunk + 1 - 4) * 32 * HW);
      __builtin_prefetch(nsrc + (size_t)(hi * 16) * HW + px0, 0, 3);
    }
    v16h bf0 = bfrag_f32(src, HW, px0, hi);
    v16h bf1 = bfrag_f32(src, HW, px1, hi);
#pragma unroll
    for (int m = 0; m < 4; ++m) {
      v16h a = *(const v16h*)(wls + ((chunk * 4 + m) * 32 + lane) * 16);
      acc0[m] = wmma(a, bf0, acc0[m]);
      acc1[m] = wmma(a, bf1, acc1[m]);
    }
  }
  float* ob0 = out + (size_t)b * 64 * HW + px0;
  float* ob1 = out + (size_t)b * 64 * HW + px1;
#pragma unroll
  for (int m = 0; m < 4; ++m)
#pragma unroll
    for (int r = 0; r < 8; ++r) {
      int o = m * 16 + r + 8 * hi;
      ob0[(size_t)o * HW] = fmaxf(acc0[m][r] + bls[o], 0.f);
      ob1[(size_t)o * HW] = fmaxf(acc1[m][r] + bls[o], 0.f);
    }
}

// ===========================================================================
// Kernel B: event path, ONLY at ::4 pixels.
//   h0  = ReLU(We0[64x128]*[event;last_event] + be0)
//   efd = ReLU(We1[64x64]*h0 + be1)  -> efd16 (f16 [b][c][i]) and ef_up.
// ===========================================================================
__global__ void __launch_bounds__(128)
k_event(const float* __restrict__ ev, const float* __restrict__ lev,
        const float* __restrict__ we0, const float* __restrict__ be0,
        const float* __restrict__ we1, const float* __restrict__ be1,
        _Float16* __restrict__ efd16, float* __restrict__ efup) {
  __shared__ __align__(32) _Float16 w0s[4 * 4 * 32 * 16];  // 8192
  __shared__ __align__(32) _Float16 w1s[2 * 4 * 32 * 16];  // 4096
  __shared__ __align__(32) _Float16 hb[4 * 64 * 16];       // per-wave h0
  __shared__ float b0s[64], b1s[64];
  const int tid = threadIdx.x;
  for (int t = tid; t < 64 * 128; t += 128)
    w0s[wswz(t / 128, t % 128)] = (_Float16)we0[t];
  for (int t = tid; t < 64 * 64; t += 128)
    w1s[wswz(t / 64, t % 64)] = (_Float16)we1[t];
  if (tid < 64) b0s[tid] = be0[tid];
  else b1s[tid - 64] = be1[tid - 64];
  __syncthreads();

  const int lane = tid & 31, wv = tid >> 5;
  const int n = lane & 15, hi = lane >> 4;
  const int b = blockIdx.y;
  const int i0 = blockIdx.x * 64 + wv * 16;
  const int idx = i0 + n;                 // downsampled flat index (hd*64+wd)
  const int hd = idx >> 6, wd = idx & 63;
  const size_t px = (size_t)hd * 1024 + (size_t)wd * 4;  // (4hd)*256 + 4wd

  // stage 1 (K = 128)
  v8f a1[4] = {v8f_zero(), v8f_zero(), v8f_zero(), v8f_zero()};
#pragma unroll
  for (int chunk = 0; chunk < 4; ++chunk) {
    const float* src = ((chunk < 2) ? ev : lev) + (size_t)b * 64 * HW +
                       (size_t)(chunk & 1) * 32 * HW;
    v16h bf = bfrag_f32(src, HW, px, hi);
#pragma unroll
    for (int m = 0; m < 4; ++m) {
      v16h a = *(const v16h*)(w0s + ((chunk * 4 + m) * 32 + lane) * 16);
      a1[m] = wmma(a, bf, a1[m]);
    }
  }
  _Float16* hbw = hb + wv * (64 * 16);
#pragma unroll
  for (int m = 0; m < 4; ++m)
#pragma unroll
    for (int r = 0; r < 8; ++r) {
      int o = m * 16 + r + 8 * hi;
      hbw[o * 16 + n] = (_Float16)fmaxf(a1[m][r] + b0s[o], 0.f);
    }
  asm volatile("s_wait_dscnt 0x0" ::: "memory");  // same-wave LDS RAW

  // stage 2 (K = 64)
  v8f a2[4] = {v8f_zero(), v8f_zero(), v8f_zero(), v8f_zero()};
#pragma unroll
  for (int chunk = 0; chunk < 2; ++chunk) {
    v16h bf = bfrag_lds(hbw, chunk * 32, n, hi);
#pragma unroll
    for (int m = 0; m < 4; ++m) {
      v16h a = *(const v16h*)(w1s + ((chunk * 4 + m) * 32 + lane) * 16);
      a2[m] = wmma(a, bf, a2[m]);
    }
  }
#pragma unroll
  for (int m = 0; m < 4; ++m)
#pragma unroll
    for (int r = 0; r < 8; ++r) {
      int o = m * 16 + r + 8 * hi;
      float e = fmaxf(a2[m][r] + b1s[o], 0.f);
      efd16[((size_t)b * 64 + o) * ND + idx] = (_Float16)e;
      float4 e4 = make_float4(e, e, e, e);
      float* ob = efup + ((size_t)b * 64 + o) * HW + (size_t)hd * 1024 + wd * 4;
#pragma unroll
      for (int dy = 0; dy < 4; ++dy) *(float4*)(ob + dy * 256) = e4;
    }
}

// ===========================================================================
// Kernel C: x2d = ReLU(Wx1*x_ds+b); q = ReLU(Wq*x2d+b); v = ReLU(Wv*x2d+b);
//           k = ReLU(Wk*efd+b).  q,k -> f16 [b][c][i];  v -> f16 [b][i][c].
// ===========================================================================
__global__ void __launch_bounds__(128)
k_proj(const float* __restrict__ xfull, const _Float16* __restrict__ efd16,
       const float* __restrict__ wx1, const float* __restrict__ bx1,
       const float* __restrict__ wq, const float* __restrict__ bq,
       const float* __restrict__ wk, const float* __restrict__ bk,
       const float* __restrict__ wv, const float* __restrict__ bv,
       _Float16* __restrict__ q16, _Float16* __restrict__ k16,
       _Float16* __restrict__ vT16) {
  __shared__ __align__(32) _Float16 wxs[4096], wqs[4096], wks[4096], wvs[4096];
  __shared__ __align__(32) _Float16 xb[4 * 64 * 16];
  __shared__ float bxs[64], bqs[64], bks[64], bvs[64];
  const int tid = threadIdx.x;
  for (int t = tid; t < 4096; t += 128) {
    int o = t >> 6, i = t & 63, s = wswz(o, i);
    wxs[s] = (_Float16)wx1[t]; wqs[s] = (_Float16)wq[t];
    wks[s] = (_Float16)wk[t];  wvs[s] = (_Float16)wv[t];
  }
  if (tid < 64) { bxs[tid] = bx1[tid]; bqs[tid] = bq[tid]; }
  else { bks[tid - 64] = bk[tid - 64]; bvs[tid - 64] = bv[tid - 64]; }
  __syncthreads();

  const int lane = tid & 31, wv_ = tid >> 5;
  const int n = lane & 15, hi = lane >> 4;
  const int b = blockIdx.y;
  const int i0 = blockIdx.x * 64 + wv_ * 16;
  const int idx = i0 + n;
  const int hd = idx >> 6, wd = idx & 63;
  const size_t px = (size_t)hd * 1024 + (size_t)wd * 4;

  // x2d
  v8f ax[4] = {v8f_zero(), v8f_zero(), v8f_zero(), v8f_zero()};
#pragma unroll
  for (int chunk = 0; chunk < 2; ++chunk) {
    const float* src = xfull + (size_t)b * 64 * HW + (size_t)chunk * 32 * HW;
    v16h bf = bfrag_f32(src, HW, px, hi);
#pragma unroll
    for (int m = 0; m < 4; ++m) {
      v16h a = *(const v16h*)(wxs + ((chunk * 4 + m) * 32 + lane) * 16);
      ax[m] = wmma(a, bf, ax[m]);
    }
  }
  _Float16* xbw = xb + wv_ * (64 * 16);
#pragma unroll
  for (int m = 0; m < 4; ++m)
#pragma unroll
    for (int r = 0; r < 8; ++r) {
      int o = m * 16 + r + 8 * hi;
      xbw[o * 16 + n] = (_Float16)fmaxf(ax[m][r] + bxs[o], 0.f);
    }
  asm volatile("s_wait_dscnt 0x0" ::: "memory");

  // q and v share the B fragments built from x2d
  v8f aq[4] = {v8f_zero(), v8f_zero(), v8f_zero(), v8f_zero()};
  v8f av[4] = {v8f_zero(), v8f_zero(), v8f_zero(), v8f_zero()};
#pragma unroll
  for (int chunk = 0; chunk < 2; ++chunk) {
    v16h bf = bfrag_lds(xbw, chunk * 32, n, hi);
#pragma unroll
    for (int m = 0; m < 4; ++m) {
      v16h a = *(const v16h*)(wqs + ((chunk * 4 + m) * 32 + lane) * 16);
      aq[m] = wmma(a, bf, aq[m]);
      v16h a2 = *(const v16h*)(wvs + ((chunk * 4 + m) * 32 + lane) * 16);
      av[m] = wmma(a2, bf, av[m]);
    }
  }
  // k from efd16 (f16 [c][i], contiguous in i)
  v8f ak[4] = {v8f_zero(), v8f_zero(), v8f_zero(), v8f_zero()};
  const _Float16* eb = efd16 + (size_t)b * 64 * ND;
#pragma unroll
  for (int chunk = 0; chunk < 2; ++chunk) {
    v16h bf = bfrag_f16(eb + (size_t)chunk * 32 * ND, ND, idx, hi);
#pragma unroll
    for (int m = 0; m < 4; ++m) {
      v16h a = *(const v16h*)(wks + ((chunk * 4 + m) * 32 + lane) * 16);
      ak[m] = wmma(a, bf, ak[m]);
    }
  }
#pragma unroll
  for (int m = 0; m < 4; ++m)
#pragma unroll
    for (int r = 0; r < 8; ++r) {
      int o = m * 16 + r + 8 * hi;
      q16[((size_t)b * 64 + o) * ND + idx] = (_Float16)fmaxf(aq[m][r] + bqs[o], 0.f);
      k16[((size_t)b * 64 + o) * ND + idx] = (_Float16)fmaxf(ak[m][r] + bks[o], 0.f);
      vT16[((size_t)b * ND + idx) * 64 + o] = (_Float16)fmaxf(av[m][r] + bvs[o], 0.f);
    }
}

// ===========================================================================
// Kernel D: flash attention per batch.  S = Qt(16x64) x K(64x32j);
// online softmax; oT(16x64) += P(16x32) x Vt(32x64).  oT f32 [b][i][c].
// ===========================================================================
__global__ void __launch_bounds__(128)
k_attn(const _Float16* __restrict__ q16, const _Float16* __restrict__ k16,
       const _Float16* __restrict__ vT16, float* __restrict__ oT) {
  __shared__ __align__(32) _Float16 pbuf[4 * 16 * 32];
  const int tid = threadIdx.x;
  const int lane = tid & 31, wv = tid >> 5;
  const int n = lane & 15, hi = lane >> 4;
  const int b = blockIdx.y;
  const int i0 = blockIdx.x * 64 + wv * 16;
  const _Float16* qb = q16 + (size_t)b * 64 * ND;
  const _Float16* kb = k16 + (size_t)b * 64 * ND;
  const _Float16* vb = vT16 + (size_t)b * ND * 64;
  _Float16* pw = pbuf + wv * 512;

  // A fragments of Q block (reused for all 4096 keys)
  v16h aqf[2];
#pragma unroll
  for (int chunk = 0; chunk < 2; ++chunk)
#pragma unroll
    for (int h = 0; h < 16; ++h) {
      int c = chunk * 32 + (h >> 3) * 16 + hi * 8 + (h & 7);
      aqf[chunk][h] = qb[(size_t)c * ND + i0 + (lane & 15)];
    }

  v8f acc[4] = {v8f_zero(), v8f_zero(), v8f_zero(), v8f_zero()};
  float mrow[8], lrow[8];
#pragma unroll
  for (int r = 0; r < 8; ++r) { mrow[r] = -1e30f; lrow[r] = 0.f; }

  for (int j0 = 0; j0 < ND; j0 += 32) {
    v8f s0 = v8f_zero(), s1 = v8f_zero();
#pragma unroll
    for (int chunk = 0; chunk < 2; ++chunk) {
      const _Float16* kc = kb + (size_t)chunk * 32 * ND;
      v16h bk0 = bfrag_f16(kc, ND, (size_t)j0 + n, hi);
      v16h bk1 = bfrag_f16(kc, ND, (size_t)j0 + 16 + n, hi);
      s0 = wmma(aqf[chunk], bk0, s0);
      s1 = wmma(aqf[chunk], bk1, s1);
    }
#pragma unroll
    for (int r = 0; r < 8; ++r) {
      float rm = fmaxf(s0[r], s1[r]);
      rm = fmaxf(rm, __shfl_xor(rm, 1, 32));
      rm = fmaxf(rm, __shfl_xor(rm, 2, 32));
      rm = fmaxf(rm, __shfl_xor(rm, 4, 32));
      rm = fmaxf(rm, __shfl_xor(rm, 8, 32));
      float mn = fmaxf(mrow[r], rm);
      float sc = __expf(mrow[r] - mn);
      lrow[r] *= sc;
      acc[0][r] *= sc; acc[1][r] *= sc; acc[2][r] *= sc; acc[3][r] *= sc;
      mrow[r] = mn;
      float p0 = __expf(s0[r] - mn), p1 = __expf(s1[r] - mn);
      float rs = p0 + p1;
      rs += __shfl_xor(rs, 1, 32);
      rs += __shfl_xor(rs, 2, 32);
      rs += __shfl_xor(rs, 4, 32);
      rs += __shfl_xor(rs, 8, 32);
      lrow[r] += rs;
      int irow = r + 8 * hi;
      pw[irow * 32 + n] = (_Float16)p0;
      pw[irow * 32 + 16 + n] = (_Float16)p1;
    }
    asm volatile("s_wait_dscnt 0x0" ::: "memory");
    v16h ap;
#pragma unroll
    for (int h = 0; h < 16; ++h) {
      int jj = (h >> 3) * 16 + hi * 8 + (h & 7);
      ap[h] = pw[(lane & 15) * 32 + jj];
    }
#pragma unroll
    for (int ct = 0; ct < 4; ++ct) {
      v16h bv_ = bfrag_f16(vb + (size_t)j0 * 64, 64, (size_t)ct * 16 + n, hi);
      acc[ct] = wmma(ap, bv_, acc[ct]);
    }
  }
#pragma unroll
  for (int r = 0; r < 8; ++r) {
    float inv = 1.f / lrow[r];
    float* ob = oT + ((size_t)b * ND + i0 + r + 8 * hi) * 64;
#pragma unroll
    for (int ct = 0; ct < 4; ++ct) ob[ct * 16 + n] = acc[ct][r] * inv;
  }
}

// ===========================================================================
// Kernel E: out = x + THITA * gamma * up4(o)  (in-place on d_out region 0)
// ===========================================================================
__global__ void __launch_bounds__(256)
k_final(const float* __restrict__ oT, const float* __restrict__ gamma,
        float* __restrict__ out) {
  int idx = blockIdx.x * 256 + threadIdx.x;
  float g = gamma[0] * THITA;
  int p = idx & (HW - 1);
  int bc = idx >> 16;
  int b = bc >> 6, c = bc & 63;
  int h = p >> 8, w = p & 255;
  int i = (h >> 2) * 64 + (w >> 2);
  out[idx] = out[idx] + g * oT[((size_t)b * ND + i) * 64 + c];
}

extern "C" void kernel_launch(void* const* d_in, const int* in_sizes, int n_in,
                              void* d_out, int out_size, void* d_ws,
                              size_t ws_size, hipStream_t stream) {
  const float* x1  = (const float*)d_in[0];
  const float* x2  = (const float*)d_in[1];
  const float* ev  = (const float*)d_in[2];
  const float* lev = (const float*)d_in[3];
  const float* wx0 = (const float*)d_in[4];
  const float* bx0 = (const float*)d_in[5];
  const float* wx1 = (const float*)d_in[6];
  const float* bx1 = (const float*)d_in[7];
  const float* we0 = (const float*)d_in[8];
  const float* be0 = (const float*)d_in[9];
  const float* we1 = (const float*)d_in[10];
  const float* be1 = (const float*)d_in[11];
  const float* wq  = (const float*)d_in[12];
  const float* bq  = (const float*)d_in[13];
  const float* wk  = (const float*)d_in[14];
  const float* bk  = (const float*)d_in[15];
  const float* wv  = (const float*)d_in[16];
  const float* bv  = (const float*)d_in[17];
  const float* gamma = (const float*)d_in[18];

  float* out_x  = (float*)d_out;                                // [B,C,H,W]
  float* out_ef = (float*)d_out + (size_t)NB * NC * HW;         // ef_up

  char* w = (char*)d_ws;
  const size_t SZ16 = (size_t)NB * NC * ND * sizeof(_Float16);  // 2 MB
  _Float16* efd16 = (_Float16*)(w);
  _Float16* q16   = (_Float16*)(w + SZ16);
  _Float16* k16   = (_Float16*)(w + 2 * SZ16);
  _Float16* vT16  = (_Float16*)(w + 3 * SZ16);
  float*    oT    = (float*)(w + 4 * SZ16);                     // 4 MB

  // A: full-res fused conv -> x (residual) into d_out (32 px per wave)
  k_convx<<<dim3(HW / 128, NB), 128, 0, stream>>>(x1, x2, wx0, bx0, out_x);
  // B: event path at ::4 only -> efd16 + ef_up
  k_event<<<dim3(ND / 64, NB), 128, 0, stream>>>(ev, lev, we0, be0, we1, be1,
                                                 efd16, out_ef);
  // C: x2d -> q,v ; efd -> k
  k_proj<<<dim3(ND / 64, NB), 128, 0, stream>>>(out_x, efd16, wx1, bx1, wq, bq,
                                                wk, bk, wv, bv, q16, k16, vT16);
  // D: flash attention
  k_attn<<<dim3(ND / 64, NB), 128, 0, stream>>>(q16, k16, vT16, oT);
  // E: residual + upsampled attention output (in place on d_out)
  k_final<<<dim3((NB * NC * HW) / 256), 256, 0, stream>>>(oT, gamma, out_x);
}
